// ScaledDotProduct_15839839388402
// MI455X (gfx1250) — matrix operand order repaired and checked by reading, not compiled, not run on hardware
//
#include <hip/hip_runtime.h>

typedef __attribute__((ext_vector_type(16))) _Float16 v16h;
typedef __attribute__((ext_vector_type(8)))  _Float16 v8h;
typedef __attribute__((ext_vector_type(8)))  float    v8f;
typedef __attribute__((ext_vector_type(4)))  float    v4f;

#define TGT 2048
#define SRC 2048
#define BH  16
#define D   64
#define ROWS 32          // tgt rows per workgroup
#define NTHREADS 256     // 8 waves (wave32)
#define SPAD (SRC + 4)   // pad logits rows: keeps b128 alignment, de-phases banks

// ---------------------------------------------------------------------------
// Prep: K [s][bh][d] f32 -> Kh [bh][s][d] f16 (coalesced), and
//       V [s][bh][d] f32 -> Vt [bh][d][s] f16 via LDS-tiled 64x64 transpose
// so both writes are contiguous 128B runs (no 2-byte scatter).
// ---------------------------------------------------------------------------
__global__ __launch_bounds__(256)
void prep_kernel(const float* __restrict__ K, const float* __restrict__ V,
                 _Float16* __restrict__ Kh, _Float16* __restrict__ Vt)
{
    __shared__ float vtile[64][65];            // pad 65: stride 65 % 64 == 1
    const int s0  = blockIdx.x * 64;
    const int bh  = blockIdx.y;
    const int tid = threadIdx.x;

    for (int i = tid; i < 64 * 64; i += 256) {
        const int sl = i >> 6, d = i & 63;
        const size_t gi = ((size_t)(s0 + sl) * BH + bh) * D + d;
        Kh[((size_t)bh * SRC + (s0 + sl)) * D + d] = (_Float16)K[gi];
        vtile[sl][d] = V[gi];
    }
    __syncthreads();
    for (int i = tid; i < 64 * 64; i += 256) {
        const int d = i >> 6, sl = i & 63;     // consecutive tid -> consecutive s
        Vt[((size_t)bh * D + d) * SRC + (s0 + sl)] = (_Float16)vtile[sl][d];
    }
}

// ---------------------------------------------------------------------------
// Fused attention: S = scale*Q.K^T -> vectorized mask pass -> exact softmax in
// 263 KB LDS -> one non-temporal streaming write of w (268 MB) -> out = P.V.
// ---------------------------------------------------------------------------
__global__ __launch_bounds__(NTHREADS)
void attn_fused_kernel(const float* __restrict__ Q,
                       const _Float16* __restrict__ Kh,  // [BH][SRC][D]
                       const _Float16* __restrict__ Vt,  // [BH][D][SRC]
                       const unsigned char* __restrict__ mask,
                       float* __restrict__ out,          // [TGT, BH, D]
                       float* __restrict__ wout)         // [BH, TGT, SRC]
{
    __shared__ float     logits[ROWS][SPAD];   // ~263 KB (CDNA5: 320 KB/WGP)
    __shared__ _Float16  qsh[ROWS][D];
    __shared__ float     red[ROWS][8];
    __shared__ float     rowinv[ROWS];

    const int bh   = blockIdx.y;
    const int m0   = blockIdx.x * ROWS;
    const int tid  = threadIdx.x;
    const int lane = tid & 31;
    const int wave = tid >> 5;

    const int rA = lane & 15;      // A row (M) / B,C column (N) selector
    const int cA = lane >> 4;      // lane half selector
    const int mt = wave >> 2;      // which 16-row half of the 32-row tile
    const int wq = wave & 3;       // which 16-col quad

    // ---- Phase 0: Q tile -> f16 LDS, pre-scaled by 1/sqrt(D) ----------------
    for (int i = tid; i < ROWS * D; i += NTHREADS) {
        const int r = i >> 6, d = i & (D - 1);
        qsh[r][d] = (_Float16)(Q[((size_t)(m0 + r) * BH + bh) * D + d] * 0.125f);
    }
    __syncthreads();

    // ---- Phase 1: S = Qs . K^T into LDS (pure load -> wmma -> store) --------
    for (int n0 = 0; n0 < SRC; n0 += 64) {
        const int ncol = n0 + wq * 16 + rA;            // src column (B & C)
        v8f acc = {};
        #pragma unroll
        for (int kb = 0; kb < D; kb += 32) {
            v8h alo = *(const v8h*)&qsh[mt * 16 + rA][kb + cA * 8];
            v8h ahi = *(const v8h*)&qsh[mt * 16 + rA][kb + 16 + cA * 8];
            v16h a = __builtin_shufflevector(alo, ahi,
                        0,1,2,3,4,5,6,7,8,9,10,11,12,13,14,15);
            const _Float16* kp = &Kh[((size_t)bh * SRC + ncol) * D + kb + cA * 8];
            v8h blo = *(const v8h*)kp;
            v8h bhi = *(const v8h*)(kp + 16);
            v16h b = __builtin_shufflevector(blo, bhi,
                        0,1,2,3,4,5,6,7,8,9,10,11,12,13,14,15);
            acc = __builtin_amdgcn_wmma_f32_16x16x32_f16(
                      false, a, false, b, (short)0, acc, false, false);
        }
        const int rb = mt * 16 + cA * 8;               // C row base
        #pragma unroll
        for (int v = 0; v < 8; ++v) logits[rb + v][ncol] = acc[v];
    }
    __syncthreads();

    // ---- Phase 1b: vectorized mask pass -------------------------------------
    // 16 mask bytes per b128 load (coalesced), 16 logits per 4x ds b128 RMW.
    for (int idx = tid; idx < (ROWS * SRC) / 16; idx += NTHREADS) {
        const int base = idx << 4;
        const int r = base >> 11;
        const int c = base & (SRC - 1);
        const uint4 mv = *(const uint4*)&mask[(size_t)(m0 + r) * SRC + c];
        float* lp = &logits[r][c];
        v4f x0 = *(v4f*)lp;
        v4f x1 = *(v4f*)(lp + 4);
        v4f x2 = *(v4f*)(lp + 8);
        v4f x3 = *(v4f*)(lp + 12);
        #pragma unroll
        for (int e = 0; e < 4; ++e) {
            if ((mv.x >> (e * 8)) & 0xffu) x0[e] = -1e8f;
            if ((mv.y >> (e * 8)) & 0xffu) x1[e] = -1e8f;
            if ((mv.z >> (e * 8)) & 0xffu) x2[e] = -1e8f;
            if ((mv.w >> (e * 8)) & 0xffu) x3[e] = -1e8f;
        }
        *(v4f*)lp        = x0;
        *(v4f*)(lp + 4)  = x1;
        *(v4f*)(lp + 8)  = x2;
        *(v4f*)(lp + 12) = x3;
    }
    __syncthreads();

    // ---- Phase 2: exact softmax over each 2048-wide row ---------------------
    {
        const int r = tid >> 3, part = tid & 7;        // 8 threads per row
        float mx = -3.0e38f;
        for (int cc = 0; cc < 256; ++cc) {
            const int c = part + (cc << 3);            // interleave banks
            mx = fmaxf(mx, logits[r][c]);
        }
        red[r][part] = mx;
        __syncthreads();
        float rmx = red[r][0];
        #pragma unroll
        for (int p = 1; p < 8; ++p) rmx = fmaxf(rmx, red[r][p]);
        __syncthreads();
        float sum = 0.0f;
        for (int cc = 0; cc < 256; ++cc) {
            const int c = part + (cc << 3);
            const float e = __expf(logits[r][c] - rmx);
            logits[r][c] = e;
            sum += e;
        }
        red[r][part] = sum;
        __syncthreads();
        float tot = red[r][0];
        #pragma unroll
        for (int p = 1; p < 8; ++p) tot += red[r][p];
        if (part == 0) rowinv[r] = 1.0f / tot;
    }
    __syncthreads();

    // normalize + single coalesced non-temporal streaming write of w (268 MB)
    for (int i = tid; i < ROWS * SRC; i += NTHREADS) {
        const int r = i >> 11, c = i & (SRC - 1);
        const float p = logits[r][c] * rowinv[r];
        logits[r][c] = p;
        __builtin_nontemporal_store(
            p, &wout[((size_t)bh * TGT + (m0 + r)) * SRC + c]);
    }
    __syncthreads();

    // ---- Phase 3: out = P . V over the 2048-long src axis -------------------
    const int ar   = mt * 16 + rA;                 // A-fragment row
    const int dcol = wq * 16 + rA;                 // head-dim column (B & C)
    v8f acc2 = {};
    for (int kk = 0; kk < SRC; kk += 32) {
        const float* lp = &logits[ar][kk + cA * 8];
        v4f p0 = *(const v4f*)lp;
        v4f p1 = *(const v4f*)(lp + 4);
        v4f p2 = *(const v4f*)(lp + 16);
        v4f p3 = *(const v4f*)(lp + 20);
        v16h a;
        #pragma unroll
        for (int e = 0; e < 4; ++e) {
            a[e]      = (_Float16)p0[e];
            a[e + 4]  = (_Float16)p1[e];
            a[e + 8]  = (_Float16)p2[e];
            a[e + 12] = (_Float16)p3[e];
        }
        const _Float16* vp = &Vt[((size_t)bh * D + dcol) * SRC + kk + cA * 8];
        v8h blo = *(const v8h*)vp;
        v8h bhi = *(const v8h*)(vp + 16);
        v16h b = __builtin_shufflevector(blo, bhi,
                    0,1,2,3,4,5,6,7,8,9,10,11,12,13,14,15);
        acc2 = __builtin_amdgcn_wmma_f32_16x16x32_f16(
                   false, a, false, b, (short)0, acc2, false, false);
    }
    const int rb = mt * 16 + cA * 8;
    #pragma unroll
    for (int v = 0; v < 8; ++v) {
        out[((size_t)(m0 + rb + v) * BH + bh) * D + dcol] = acc2[v];
    }
}

extern "C" void kernel_launch(void* const* d_in, const int* in_sizes, int n_in,
                              void* d_out, int out_size, void* d_ws, size_t ws_size,
                              hipStream_t stream) {
    const float* Q = (const float*)d_in[0];
    const float* K = (const float*)d_in[1];
    const float* V = (const float*)d_in[2];
    const unsigned char* mask = (const unsigned char*)d_in[3];
    float* out  = (float*)d_out;                       // [TGT, BH, D]
    float* wout = out + (size_t)TGT * BH * D;          // [BH, TGT, SRC]

    _Float16* Kh = (_Float16*)d_ws;                    // [BH][SRC][D]  4 MB
    _Float16* Vt = Kh + (size_t)BH * SRC * D;          // [BH][D][SRC]  4 MB

    dim3 pgrid(SRC / 64, BH);
    prep_kernel<<<pgrid, 256, 0, stream>>>(K, V, Kh, Vt);

    dim3 grid(TGT / ROWS, BH);
    attn_fused_kernel<<<grid, NTHREADS, 0, stream>>>(Q, Kh, Vt, mask, out, wout);
}
